// KGPool_17197049053680
// MI455X (gfx1250) — compile-verified, matrix operand order checked by reading, not run on hardware
//
#include <hip/hip_runtime.h>
#include <math.h>

typedef __attribute__((ext_vector_type(16))) _Float16 v16h;
typedef __attribute__((ext_vector_type(8)))  float    v8f;
typedef __attribute__((ext_vector_type(2)))  float    v2f;

#define NEG_INF (-3.402823466e38f)

#if __has_builtin(__builtin_amdgcn_wmma_f32_16x16x4_f32)
#define KG_USE_F32_WMMA 1
#else
#define KG_USE_F32_WMMA 0
#endif

// ---------------------------------------------------------------------------
// Kernel 1: h = x @ W   (GEMV padded into 16x16 WMMA tiles)
// One wave handles 16 node rows. B = W broadcast into all 16 columns so every
// column of D carries the score; lanes 0/16 extract 8 rows each.
// Preferred path: full-precision V_WMMA_F32_16X16X4_F32 (matches reference f32
// math). Fallback: V_WMMA_F32_16X16X32_F16 (codegen-verified).
// ---------------------------------------------------------------------------
__global__ void k_gemv_wmma(const float* __restrict__ x, const float* __restrict__ W,
                            float* __restrict__ h, int N, int C) {
    const int lane = threadIdx.x & 31;
    const int wave = (blockIdx.x * blockDim.x + threadIdx.x) >> 5;
    const int base = wave * 16;
    if (base >= N) return;                 // uniform per wave: EXEC stays all-1s
    const int m  = lane & 15;
    const int hi = lane >> 4;
    int row = base + m;
    if (row >= N) row = N - 1;             // clamp (duplicates computed, not stored)
    const float* __restrict__ xr = x + (size_t)row * (size_t)C;

    v8f acc = {};
#if KG_USE_F32_WMMA
    // 32-bit A 16x4 layout: lane m (lanes 0-15) K={0,1}; lanes 16-31 K={2,3}
    for (int kb = 0; kb + 4 <= C; kb += 4) {
        const int k0 = kb + 2 * hi;
        v2f a, bm;
        a[0]  = xr[k0];     a[1]  = xr[k0 + 1];
        bm[0] = W[k0];      bm[1] = W[k0 + 1];   // all 16 B-columns identical
        acc = __builtin_amdgcn_wmma_f32_16x16x4_f32(
            /*neg_a=*/false, a, /*neg_b=*/false, bm,
            /*c_mod=*/(short)0, acc, /*reuse_a=*/false, /*reuse_b=*/false);
    }
#else
    for (int kb = 0; kb + 32 <= C; kb += 32) {
        v16h a, bm;
        const int ka = kb + 8 * hi;        // A 16-bit layout: K = 8*hi + {0..7, 16..23}
#pragma unroll
        for (int j = 0; j < 8; ++j) {
            a[j]     = (_Float16)xr[ka + j];
            a[j + 8] = (_Float16)xr[ka + 16 + j];
        }
        const int kw = kb + 16 * hi;       // B: lane-half holds K = 16*hi + j
#pragma unroll
        for (int j = 0; j < 16; ++j) bm[j] = (_Float16)W[kw + j];
        acc = __builtin_amdgcn_wmma_f32_16x16x32_f16(
            false, a, false, bm, (short)0, acc, false, false);
    }
#endif
    if (m == 0) {                          // lanes 0 (rows base+0..7) and 16 (base+8..15)
#pragma unroll
        for (int r = 0; r < 8; ++r) {
            int orow = base + 8 * hi + r;
            if (orow < N) h[orow] = acc[r];
        }
    }
}

// ---------------------------------------------------------------------------
// Init: deg = 1 (self-loop), agg = 0
// ---------------------------------------------------------------------------
__global__ void k_init(float* __restrict__ deg, float* __restrict__ agg, int N) {
    int i = blockIdx.x * blockDim.x + threadIdx.x;
    if (i < N) { deg[i] = 1.0f; agg[i] = 0.0f; }
}

__global__ void k_degree(const int* __restrict__ col, float* __restrict__ deg, int E) {
    int e = blockIdx.x * blockDim.x + threadIdx.x;
    if (e < E) atomicAdd(&deg[col[e]], 1.0f);
}

__global__ void k_dis(const float* __restrict__ deg, float* __restrict__ dis, int N) {
    int i = blockIdx.x * blockDim.x + threadIdx.x;
    if (i < N) dis[i] = rsqrtf(deg[i]);
}

__global__ void k_aggregate(const int* __restrict__ row, const int* __restrict__ col,
                            const float* __restrict__ h, const float* __restrict__ dis,
                            float* __restrict__ agg, int E) {
    int e = blockIdx.x * blockDim.x + threadIdx.x;
    if (e < E) {
        int r = row[e], c = col[e];
        atomicAdd(&agg[c], h[r] * dis[r] * dis[c]);
    }
}

__global__ void k_finalscore(const float* __restrict__ agg, const float* __restrict__ h,
                             const float* __restrict__ dis, const float* __restrict__ b,
                             float* __restrict__ score, int N) {
    int i = blockIdx.x * blockDim.x + threadIdx.x;
    if (i < N) score[i] = agg[i] + h[i] * dis[i] * dis[i] + b[0];
}

// ---------------------------------------------------------------------------
// Per-sentence fused softmax + std cutoff + keep mask. One 64-thread WG per
// contiguous segment [starts[s], starts[s+1]).
// ---------------------------------------------------------------------------
__global__ void k_sentence(const float* __restrict__ score, const int* __restrict__ starts,
                           int* __restrict__ keep, int N, int S,
                           const int* __restrict__ alpha_p) {
    const int s = blockIdx.x;
    const int begin = (s == 0) ? 0 : starts[s];
    const int end   = (s == S - 1) ? N : starts[s + 1];
    if (begin >= end) return;
    const float alpha = (float)alpha_p[0];
    const int t = threadIdx.x;
    __shared__ float sm[64];

    // pass 1: segment max
    float mx = NEG_INF;
    for (int i = begin + t; i < end; i += 64) mx = fmaxf(mx, score[i]);
    sm[t] = mx; __syncthreads();
    for (int off = 32; off > 0; off >>= 1) {
        if (t < off) sm[t] = fmaxf(sm[t], sm[t + off]);
        __syncthreads();
    }
    const float mseg = sm[0];
    __syncthreads();

    // pass 2: Z = sum e, ZZ = sum e^2
    float z = 0.0f, z2 = 0.0f;
    for (int i = begin + t; i < end; i += 64) {
        float e = expf(score[i] - mseg);
        z += e; z2 += e * e;
    }
    sm[t] = z; __syncthreads();
    for (int off = 32; off > 0; off >>= 1) { if (t < off) sm[t] += sm[t + off]; __syncthreads(); }
    const float Z = sm[0];
    __syncthreads();
    sm[t] = z2; __syncthreads();
    for (int off = 32; off > 0; off >>= 1) { if (t < off) sm[t] += sm[t + off]; __syncthreads(); }
    const float ZZ = sm[0];
    __syncthreads();

    const float cnt  = (float)(end - begin);
    const float pmax = 1.0f / Z;                       // max p = exp(0)/Z
    const float mean = 1.0f / cnt;                     // sum p == 1
    const float var  = ZZ / (Z * Z * cnt) - mean * mean;
    const float stdd = sqrtf(fmaxf(var, 0.0f));
    const float cutoff = pmax - alpha * stdd;

    // pass 3: keep mask
    for (int i = begin + t; i < end; i += 64) {
        float p = expf(score[i] - mseg) / Z;
        keep[i] = (p >= cutoff) ? 1 : 0;
    }
}

__global__ void k_union(const int* __restrict__ n1, const int* __restrict__ n2,
                        const int* __restrict__ ns, int* __restrict__ keep, int S) {
    int i = blockIdx.x * blockDim.x + threadIdx.x;
    if (i < S) { keep[n1[i]] = 1; keep[n2[i]] = 1; keep[ns[i]] = 1; }
}

// ---------------------------------------------------------------------------
// Prefix sum (pos = cumsum(keep) - 1): block sums -> scan sums -> finalize
// Block granule = 1024 elements (256 threads x 4).
// ---------------------------------------------------------------------------
__global__ void k_block_sums(const int* __restrict__ keep, int* __restrict__ bsums, int N) {
    const int t = threadIdx.x;
    const int base = blockIdx.x * 1024 + t * 4;
    int s = 0;
#pragma unroll
    for (int i = 0; i < 4; ++i) { int idx = base + i; if (idx < N) s += keep[idx]; }
    __shared__ int sm[256];
    sm[t] = s; __syncthreads();
    for (int off = 128; off > 0; off >>= 1) { if (t < off) sm[t] += sm[t + off]; __syncthreads(); }
    if (t == 0) bsums[blockIdx.x] = sm[0];
}

__global__ void k_scan_bsums(int* __restrict__ bsums, int nb) {
    __shared__ int sm[1024];
    const int t = threadIdx.x;
    int carry = 0;
    for (int base = 0; base < nb; base += 1024) {
        int v = (base + t < nb) ? bsums[base + t] : 0;
        sm[t] = v; __syncthreads();
        for (int off = 1; off < 1024; off <<= 1) {
            int add = (t >= off) ? sm[t - off] : 0;
            __syncthreads();
            sm[t] += add;
            __syncthreads();
        }
        if (base + t < nb) bsums[base + t] = sm[t] - v + carry;   // exclusive + carry
        int total = sm[1023];
        __syncthreads();
        carry += total;
    }
}

__global__ void k_scan_final(const int* __restrict__ keep, const int* __restrict__ bexcl,
                             int* __restrict__ pos, int N) {
    const int t = threadIdx.x;
    const int base = blockIdx.x * 1024 + t * 4;
    int vals[4]; int s = 0;
#pragma unroll
    for (int i = 0; i < 4; ++i) {
        int idx = base + i;
        vals[i] = (idx < N) ? keep[idx] : 0;
        s += vals[i];
    }
    __shared__ int sm[256];
    sm[t] = s; __syncthreads();
    for (int off = 1; off < 256; off <<= 1) {
        int add = (t >= off) ? sm[t - off] : 0;
        __syncthreads();
        sm[t] += add;
        __syncthreads();
    }
    int run = sm[t] - s + bexcl[blockIdx.x];      // exclusive prefix at this thread
#pragma unroll
    for (int i = 0; i < 4; ++i) {
        int idx = base + i;
        run += vals[i];
        if (idx < N) pos[idx] = run - 1;          // inclusive cumsum - 1
    }
}

// ---------------------------------------------------------------------------
// Streaming outputs
// ---------------------------------------------------------------------------
__global__ void k_xout(const float* __restrict__ x, const float* __restrict__ score,
                       const int* __restrict__ keep, float* __restrict__ xout,
                       float* __restrict__ keep_out, int N, int C) {
    const int vpr = C >> 2;                        // float4 vectors per row
    int idx = blockIdx.x * blockDim.x + threadIdx.x;
    if (idx >= N * vpr) return;
    int n = idx / vpr;
    int v = idx - n * vpr;
    float g = keep[n] ? tanhf(score[n]) : 0.0f;
    float4 xv = ((const float4*)(x + (size_t)n * (size_t)C))[v];
    float4 o; o.x = xv.x * g; o.y = xv.y * g; o.z = xv.z * g; o.w = xv.w * g;
    ((float4*)(xout + (size_t)n * (size_t)C))[v] = o;
    if (v == 0) keep_out[n] = keep[n] ? 1.0f : 0.0f;
}

__global__ void k_edges_out(const int* __restrict__ row, const int* __restrict__ col,
                            const int* __restrict__ keep, const int* __restrict__ pos,
                            float* __restrict__ new_ei, float* __restrict__ ek_out, int E) {
    int e = blockIdx.x * blockDim.x + threadIdx.x;
    if (e >= E) return;
    int r = row[e], c = col[e];
    int ek = keep[r] & keep[c];
    ek_out[e] = ek ? 1.0f : 0.0f;
    new_ei[e]     = ek ? (float)pos[r] : -1.0f;
    new_ei[E + e] = ek ? (float)pos[c] : -1.0f;
}

__global__ void k_nodes_out(const int* __restrict__ n1, const int* __restrict__ n2,
                            const int* __restrict__ ns, const int* __restrict__ pos,
                            float* __restrict__ o1, float* __restrict__ o2,
                            float* __restrict__ o3, int S) {
    int i = blockIdx.x * blockDim.x + threadIdx.x;
    if (i >= S) return;
    o1[i] = (float)pos[n1[i]];
    o2[i] = (float)pos[n2[i]];
    o3[i] = (float)pos[ns[i]];
}

// ---------------------------------------------------------------------------
extern "C" void kernel_launch(void* const* d_in, const int* in_sizes, int n_in,
                              void* d_out, int out_size, void* d_ws, size_t ws_size,
                              hipStream_t stream) {
    const float* x     = (const float*)d_in[0];
    const int*   edge  = (const int*)d_in[1];
    const int*   n1    = (const int*)d_in[2];
    const int*   n2    = (const int*)d_in[3];
    const int*   ns    = (const int*)d_in[4];
    const float* W     = (const float*)d_in[5];
    const float* b     = (const float*)d_in[6];
    const int*   alpha = (const int*)d_in[7];

    const int C = in_sizes[5];                 // W is [C,1]
    const int N = in_sizes[0] / C;
    const int E = in_sizes[1] / 2;
    const int S = in_sizes[2];

    const int* erow = edge;                    // edge_index[0]
    const int* ecol = edge + E;                // edge_index[1]

    // workspace layout (floats/ints, N each)
    float* ws    = (float*)d_ws;
    float* h     = ws;
    float* deg   = ws + (size_t)N;
    float* dis   = ws + 2 * (size_t)N;
    float* agg   = ws + 3 * (size_t)N;
    float* score = ws + 4 * (size_t)N;
    int*   keep  = (int*)(ws + 5 * (size_t)N);
    int*   pos   = (int*)(ws + 6 * (size_t)N);
    int*   bsums = (int*)(ws + 7 * (size_t)N);

    // output layout (all float32, concatenated in return order)
    float* out    = (float*)d_out;
    float* o_x    = out;
    float* o_ei   = o_x + (size_t)N * (size_t)C;
    float* o_keep = o_ei + 2 * (size_t)E;
    float* o_ek   = o_keep + (size_t)N;
    float* o_n1   = o_ek + (size_t)E;
    float* o_n2   = o_n1 + (size_t)S;
    float* o_ns   = o_n2 + (size_t)S;

    const int TB = 256;
    const int gN  = (N + TB - 1) / TB;
    const int gE  = (E + TB - 1) / TB;
    const int gS  = (S + TB - 1) / TB;
    const int nb  = (N + 1023) / 1024;

    // 1) h = x @ W via WMMA (16 rows per wave, 8 waves per block)
    {
        int waves  = (N + 15) / 16;
        int blocks = (waves + 7) / 8;
        k_gemv_wmma<<<blocks, 256, 0, stream>>>(x, W, h, N, C);
    }
    // 2) degrees + symmetric norm
    k_init<<<gN, TB, 0, stream>>>(deg, agg, N);
    k_degree<<<gE, TB, 0, stream>>>(ecol, deg, E);
    k_dis<<<gN, TB, 0, stream>>>(deg, dis, N);
    // 3) neighbor aggregation + final score
    k_aggregate<<<gE, TB, 0, stream>>>(erow, ecol, h, dis, agg, E);
    k_finalscore<<<gN, TB, 0, stream>>>(agg, h, dis, b, score, N);
    // 4) per-sentence softmax + cutoff -> keep
    k_sentence<<<S, 64, 0, stream>>>(score, n1, keep, N, S, alpha);
    k_union<<<gS, TB, 0, stream>>>(n1, n2, ns, keep, S);
    // 5) pos = cumsum(keep) - 1
    k_block_sums<<<nb, 256, 0, stream>>>(keep, bsums, N);
    k_scan_bsums<<<1, 1024, 0, stream>>>(bsums, nb);
    k_scan_final<<<nb, 256, 0, stream>>>(keep, bsums, pos, N);
    // 6) outputs
    {
        int total  = N * (C >> 2);
        int blocks = (total + TB - 1) / TB;
        k_xout<<<blocks, TB, 0, stream>>>(x, score, keep, o_x, o_keep, N, C);
    }
    k_edges_out<<<gE, TB, 0, stream>>>(erow, ecol, keep, pos, o_ei, o_ek, E);
    k_nodes_out<<<gS, TB, 0, stream>>>(n1, n2, ns, pos, o_n1, o_n2, o_ns, S);
}
